// QueryAwareGNN_9328668967068
// MI455X (gfx1250) — matrix operand order repaired and checked by reading, not compiled, not run on hardware
//
#include <hip/hip_runtime.h>

typedef _Float16 half_t;
typedef __attribute__((ext_vector_type(16))) _Float16 v16h;
typedef __attribute__((ext_vector_type(8)))  _Float16 v8h;
typedef __attribute__((ext_vector_type(8)))  float    v8f;
typedef __attribute__((ext_vector_type(4)))  float    v4f;
typedef __attribute__((ext_vector_type(4)))  unsigned int v4u;
typedef __attribute__((ext_vector_type(8)))  int      v8i;
typedef __attribute__((ext_vector_type(4)))  int      v4i;

#define DIM      128
#define NNODES   100000
#define NEDGES   1600000
#define LDS_P    136   // half pitch: 256B row + 16B TDM pad = 272B stride (bank-friendly)

// ---------------------------------------------------------------------------
// Weight convert + transpose: Wt[n*DIM + k] = (f16) W[k*DIM + n]
// ---------------------------------------------------------------------------
__global__ void wconv_kernel(const float* __restrict__ W, half_t* __restrict__ Wt) {
    int n = blockIdx.x;       // 128 blocks
    int k = threadIdx.x;      // 128 threads
    Wt[n * DIM + k] = (half_t)W[k * DIM + n];
}

// ---------------------------------------------------------------------------
// Fused GEMM:  Y[r,:] = act( X[r,:] @ W + bias (+ res[r,:]) )
//   Wt f16 [n][k] staged into LDS by the Tensor Data Mover (one DMA per block,
//   padded to a 272B row pitch for conflict-free fragment reads).
//   A fragments pulled straight from L2-resident f32 rows, cvt to f16 in regs.
//   8 waves/block, each wave: 16-row strip x 8 N-tiles, K=128 in 4 WMMA steps.
// ---------------------------------------------------------------------------
__global__ __launch_bounds__(256)
void gemm_node_kernel(const float* __restrict__ X,
                      const half_t* __restrict__ Wt,
                      const float* __restrict__ bias,
                      const float* __restrict__ res,   // may be nullptr
                      float* __restrict__ Y,
                      int nrows, int relu_out)
{
    __shared__ __align__(16) half_t Bls[128 * LDS_P];   // 34,816 B

    const int tid  = threadIdx.x;
    const int row0 = blockIdx.x * 128;

    // ---- TDM: DMA the whole 128x128 f16 weight tile into LDS (wave 0) ----
    if ((tid >> 5) == 0) {
        unsigned ldsoff = (unsigned)(uintptr_t)(&Bls[0]);          // low 32 = LDS offset
        unsigned long long ga = (unsigned long long)(uintptr_t)Wt;
        // D# group0: count=1 | lds_addr | global_addr[56:0] | type=2
        v4u g0 = { 1u,
                   ldsoff,
                   (unsigned)(ga & 0xffffffffu),
                   (unsigned)((ga >> 32) & 0x01ffffffu) | (2u << 30) };
        // D# group1: data_size=2B, pad_enable, pad_interval=64dw(256B),
        //            pad_amount=4dw(16B); dims/tile = 128x128; dim0_stride=128
        v8i g1 = { (1 << 16) | (1 << 20) | (5 << 22) | (3 << 25),
                   128 << 16,    // tensor_dim0[15:0] @ bits 63:48
                   128 << 16,    // tensor_dim1[15:0] @ bits 95:80 (hi of dim0 = 0)
                   128 << 16,    // tile_dim0 @ bits 127:112
                   128,          // tile_dim1 @ bits 143:128
                   128,          // tensor_dim0_stride[31:0] @ bits 191:160
                   0, 0 };
        v4i gz = {0, 0, 0, 0};
#if defined(__clang_major__) && (__clang_major__ >= 23)
        v8i gz8 = {0, 0, 0, 0, 0, 0, 0, 0};
        __builtin_amdgcn_tensor_load_to_lds(g0, g1, gz, gz, gz8, 0);
#else
        __builtin_amdgcn_tensor_load_to_lds(g0, g1, gz, gz, 0);
#endif
        __builtin_amdgcn_s_wait_tensorcnt(0);
    }
    __syncthreads();

    const int wave  = tid >> 5;
    const int lane  = tid & 31;
    const int khalf = lane >> 4;   // 0: K-low half of fragment, 1: K-high half
    const int lm    = lane & 15;   // row-within-strip (A) / column (B,C)
    const int m0    = wave * 16;

    // clamp tail rows (keeps EXEC all-ones around WMMA; clamped rows are
    // discarded in the epilogue anyway)
    int gr = row0 + m0 + lm;
    if (gr > nrows - 1) gr = nrows - 1;
    const float* xrow = X + (size_t)gr * DIM;

    v8f acc[8];
    #pragma unroll
    for (int nt = 0; nt < 8; ++nt) acc[nt] = (v8f){0.f,0.f,0.f,0.f,0.f,0.f,0.f,0.f};

    #pragma unroll
    for (int k0 = 0; k0 < DIM; k0 += 32) {
        // A fragment (16x32 f16): two 32B f32 chunks from L2, cvt in regs
        const float* ap = xrow + k0 + 16 * khalf;
        v8f a0 = *(const v8f*)(ap);
        v8f a1 = *(const v8f*)(ap + 8);
        v16h afrag;
        #pragma unroll
        for (int e = 0; e < 8; ++e) {
            afrag[e]     = (half_t)a0[e];
            afrag[e + 8] = (half_t)a1[e];
        }

        #pragma unroll
        for (int nt = 0; nt < 8; ++nt) {
            // B fragment (32x16 f16): 16 contiguous K per lane from padded LDS
            const half_t* bp = &Bls[(nt * 16 + lm) * LDS_P + k0 + 16 * khalf];
            v8h b0 = *(const v8h*)(bp);
            v8h b1 = *(const v8h*)(bp + 8);
            v16h bfrag = __builtin_shufflevector(b0, b1,
                             0,1,2,3,4,5,6,7, 8,9,10,11,12,13,14,15);
            acc[nt] = __builtin_amdgcn_wmma_f32_16x16x32_f16(
                          false, afrag, false, bfrag,
                          (short)0, acc[nt], false, false);
        }
    }

    // ---- epilogue: bias (+ residual) (+ relu), f32 out ----
    #pragma unroll
    for (int nt = 0; nt < 8; ++nt) {
        #pragma unroll
        for (int g = 0; g < 8; ++g) {
            int r = row0 + m0 + g + 8 * khalf;   // C layout: VGPR g -> row g (+8 hi lanes)
            int c = nt * 16 + lm;
            if (r < nrows) {
                float v = acc[nt][g] + bias[c];
                if (res)      v += res[(size_t)r * DIM + c];
                if (relu_out) v  = fmaxf(v, 0.f);
                Y[(size_t)r * DIM + c] = v;
            }
        }
    }
}

// ---------------------------------------------------------------------------
// agg = h   (float4 grid-stride copy)
// ---------------------------------------------------------------------------
__global__ void copy_kernel(const float* __restrict__ src, float* __restrict__ dst, int n4) {
    int i = blockIdx.x * blockDim.x + threadIdx.x;
    int stride = gridDim.x * blockDim.x;
    for (; i < n4; i += stride)
        ((v4f*)dst)[i] = ((const v4f*)src)[i];
}

// ---------------------------------------------------------------------------
// Edge scatter: one wave per edge; lane handles 4 features (float4 = 512B/row).
//   m = relu(h[src] + edge_emb[edge_attr]);  agg[dst] += m  (f32 atomics, L2)
// ---------------------------------------------------------------------------
__global__ __launch_bounds__(256)
void scatter_kernel(const int* __restrict__ srcI, const int* __restrict__ dstI,
                    const int* __restrict__ attr, const float* __restrict__ emb,
                    const float* __restrict__ h, float* __restrict__ agg)
{
    int gw     = (blockIdx.x * blockDim.x + threadIdx.x) >> 5;
    int lane   = threadIdx.x & 31;
    int nwaves = (gridDim.x * blockDim.x) >> 5;

    for (int e = gw; e < NEDGES; e += nwaves) {
        int s = srcI[e];
        int d = dstI[e];
        int a = attr[e];
        v4f hv = *(const v4f*)(h   + (size_t)s * DIM + lane * 4);
        v4f ev = *(const v4f*)(emb + (size_t)a * DIM + lane * 4);
        float* ap = agg + (size_t)d * DIM + lane * 4;
        #pragma unroll
        for (int j = 0; j < 4; ++j) {
            float m = fmaxf(hv[j] + ev[j], 0.f);
            __hip_atomic_fetch_add(ap + j, m, __ATOMIC_RELAXED,
                                   __HIP_MEMORY_SCOPE_AGENT);
        }
    }
}

// ---------------------------------------------------------------------------
// Output projection: y[r] = dot(h[r,:], Wout) + b_out   (one wave per row)
// ---------------------------------------------------------------------------
__global__ __launch_bounds__(256)
void outproj_kernel(const float* __restrict__ h, const float* __restrict__ Wout,
                    const float* __restrict__ b_out, float* __restrict__ y)
{
    int gw     = (blockIdx.x * blockDim.x + threadIdx.x) >> 5;
    int lane   = threadIdx.x & 31;
    int nwaves = (gridDim.x * blockDim.x) >> 5;

    for (int r = gw; r < NNODES; r += nwaves) {
        v4f hv = *(const v4f*)(h + (size_t)r * DIM + lane * 4);
        v4f wv = *(const v4f*)(Wout + lane * 4);
        float s = hv.x * wv.x + hv.y * wv.y + hv.z * wv.z + hv.w * wv.w;
        #pragma unroll
        for (int off = 16; off > 0; off >>= 1)
            s += __shfl_down(s, off, 32);
        if (lane == 0) y[r] = s + b_out[0];
    }
}

// ---------------------------------------------------------------------------
extern "C" void kernel_launch(void* const* d_in, const int* in_sizes, int n_in,
                              void* d_out, int out_size, void* d_ws, size_t ws_size,
                              hipStream_t stream)
{
    const float* x     = (const float*)d_in[0];
    const int*   ei    = (const int*)  d_in[1];   // [2, E]: src then dst
    const int*   eattr = (const int*)  d_in[2];
    const float* eemb  = (const float*)d_in[3];
    const float* Wsrc[5] = { (const float*)d_in[4],   // Win
                             (const float*)d_in[6],   // W1_0
                             (const float*)d_in[8],   // W2_0
                             (const float*)d_in[10],  // W1_1
                             (const float*)d_in[12] };// W2_1
    const float* b_in  = (const float*)d_in[5];
    const float* b1_0  = (const float*)d_in[7];
    const float* b2_0  = (const float*)d_in[9];
    const float* b1_1  = (const float*)d_in[11];
    const float* b2_1  = (const float*)d_in[13];
    const float* Wout  = (const float*)d_in[14];
    const float* b_out = (const float*)d_in[15];
    float* y = (float*)d_out;

    // workspace layout
    char* ws = (char*)d_ws;
    size_t off = 0;
    half_t* Wt[5];
    for (int i = 0; i < 5; ++i) { Wt[i] = (half_t*)(ws + off); off += (size_t)DIM * DIM * sizeof(half_t); }
    off = (off + 255) & ~(size_t)255;
    float* h   = (float*)(ws + off); off += (size_t)NNODES * DIM * sizeof(float);
    float* agg = (float*)(ws + off); off += (size_t)NNODES * DIM * sizeof(float);
    float* hid = (float*)(ws + off); off += (size_t)NNODES * DIM * sizeof(float);

    const int*   src = ei;
    const int*   dst = ei + NEDGES;
    const float* b1[2] = { b1_0, b1_1 };
    const float* b2[2] = { b2_0, b2_1 };

    // 1. weight convert/transpose to f16 [n][k]
    for (int i = 0; i < 5; ++i)
        wconv_kernel<<<DIM, DIM, 0, stream>>>(Wsrc[i], Wt[i]);

    const int GB = (NNODES + 127) / 128;   // 782 blocks

    // 2. input projection: h = x @ Win + b_in
    gemm_node_kernel<<<GB, 256, 0, stream>>>(x, Wt[0], b_in, nullptr, h, NNODES, 0);

    // 3. two GINE layers
    for (int layer = 0; layer < 2; ++layer) {
        copy_kernel<<<12500, 256, 0, stream>>>(h, agg, NNODES * DIM / 4);
        scatter_kernel<<<16384, 256, 0, stream>>>(src, dst, eattr, eemb, h, agg);
        // hid = relu(agg @ W1 + b1)
        gemm_node_kernel<<<GB, 256, 0, stream>>>(agg, Wt[1 + 2 * layer], b1[layer],
                                                 nullptr, hid, NNODES, 1);
        // h = relu(hid @ W2 + b2 + h)
        gemm_node_kernel<<<GB, 256, 0, stream>>>(hid, Wt[2 + 2 * layer], b2[layer],
                                                 h, h, NNODES, 1);
    }

    // 4. output projection
    outproj_kernel<<<4096, 256, 0, stream>>>(h, Wout, b_out, y);
}
